// VisualRelationshipGraph_23854248362790
// MI455X (gfx1250) — compile-verified, matrix operand order checked by reading, not run on hardware
//
#include <hip/hip_runtime.h>

// ---------------------------------------------------------------------------
// Problem constants (match reference)
// ---------------------------------------------------------------------------
#define BB 8
#define NN 128
#define DD 512
#define HH 8
#define DK 64
#define FFD 2048
#define LL 2
#define MN (BB * NN)   // 1024 flattened rows

typedef __attribute__((ext_vector_type(16))) __bf16 v16bf;
typedef __attribute__((ext_vector_type(8)))  float  v8f;
typedef __attribute__((ext_vector_type(4)))  float  f4;
typedef __attribute__((ext_vector_type(4)))  unsigned int u4;
typedef unsigned short bf16u;

struct Frag128 { u4 a, b; };
struct Frag16u { bf16u u[16]; };

static __device__ inline v16bf frag_cast128(Frag128 f) { return __builtin_bit_cast(v16bf, f); }
static __device__ inline v16bf frag_cast16u(Frag16u f) { return __builtin_bit_cast(v16bf, f); }

static __device__ inline bf16u f2bf(float x) {
  unsigned u = __builtin_bit_cast(unsigned, x);
  u += 0x7FFFu + ((u >> 16) & 1u);       // round-to-nearest-even
  return (bf16u)(u >> 16);
}

// A/B 16-bit fragment, row-major source (row = lane&15, K groups by lane>>4)
static __device__ inline v16bf load_frag_rm(const bf16u* base, int stride,
                                            int row0, int k0, int lane) {
  int lo = lane & 15, hi = lane >> 4;
  const bf16u* p = base + (size_t)(row0 + lo) * stride + k0 + (hi ? 8 : 0);
  Frag128 f;
  f.a = *(const u4*)(p);
  f.b = *(const u4*)(p + 16);
  return frag_cast128(f);
}

// B fragment where memory is (K x N) row-major: element [n][k] = base[k*stride + n]
static __device__ inline v16bf load_frag_cm(const bf16u* base, int stride,
                                            int n0, int k0, int lane) {
  int lo = lane & 15, hi = lane >> 4;
  int n = n0 + lo;
  int kb = k0 + (hi ? 8 : 0);
  Frag16u f;
#pragma unroll
  for (int j = 0; j < 8; ++j) {
    f.u[j]     = base[(size_t)(kb + j) * stride + n];
    f.u[j + 8] = base[(size_t)(kb + 16 + j) * stride + n];
  }
  return frag_cast16u(f);
}

static __device__ inline v8f wmma_bf16(v16bf a, v16bf b, v8f c) {
  return __builtin_amdgcn_wmma_f32_16x16x32_bf16(false, a, false, b, (short)0, c,
                                                 false, false);
}

static __device__ inline float wred_add16(float v) {
#pragma unroll
  for (int m = 1; m < 16; m <<= 1) v += __shfl_xor(v, m, 32);
  return v;
}
static __device__ inline float wred_max16(float v) {
#pragma unroll
  for (int m = 1; m < 16; m <<= 1) v = fmaxf(v, __shfl_xor(v, m, 32));
  return v;
}
static __device__ inline float wred_add32(float v) {
#pragma unroll
  for (int m = 1; m < 32; m <<= 1) v += __shfl_xor(v, m, 32);
  return v;
}

// ---------------------------------------------------------------------------
// Prep kernels
// ---------------------------------------------------------------------------
__global__ __launch_bounds__(256) void init_x_kernel(const float* __restrict__ src,
                                                     float* __restrict__ xf,
                                                     bf16u* __restrict__ xb) {
  int i = blockIdx.x * 256 + threadIdx.x;
  float v = src[i];
  xf[i] = v;
  xb[i] = f2bf(v);
}

// dst[n*K + k] = bf16(src[k*N + n])   (weight K x N -> transposed bf16 N x K)
__global__ __launch_bounds__(256) void convert_transpose_kernel(
    const float* __restrict__ src, bf16u* __restrict__ dst, int K, int N) {
  int idx = blockIdx.x * 256 + threadIdx.x;
  if (idx >= K * N) return;
  int k = idx / N, n = idx - k * N;
  dst[(size_t)n * K + k] = f2bf(src[idx]);
}

// edge_bias[b,i,j] = mean_d edge[b,i,j,d]; one wave per (b,i,j) row of 512.
// 268 MB streamed once -> non-temporal so it doesn't evict the resident
// weights/activations from the 192 MB L2.
__global__ __launch_bounds__(256) void ebias_kernel(const float* __restrict__ ef,
                                                    float* __restrict__ eb) {
  size_t rid = (size_t)blockIdx.x * 8 + (threadIdx.x >> 5);
  int lane = threadIdx.x & 31;
  const f4* p = (const f4*)(ef + rid * DD) + lane * 4;
  float s = 0.f;
#pragma unroll
  for (int t = 0; t < 4; ++t) {
    f4 a = __builtin_nontemporal_load(p + t);
    s += a[0] + a[1] + a[2] + a[3];
  }
  s = wred_add32(s);
  if (lane == 0) eb[rid] = s * (1.0f / (float)DD);
}

// adj_norm = (A + I) / max(rowsum, 1) -> bf16 for the GCN WMMA. Wave per row.
__global__ __launch_bounds__(256) void adjnorm_kernel(const float* __restrict__ adj,
                                                      bf16u* __restrict__ adjb) {
  int rid = blockIdx.x * 8 + (threadIdx.x >> 5);   // b*128 + i
  int lane = threadIdx.x & 31;
  int i = rid & (NN - 1);
  f4 a = ((const f4*)(adj + (size_t)rid * NN))[lane];
#pragma unroll
  for (int j = 0; j < 4; ++j) a[j] += ((lane * 4 + j) == i) ? 1.f : 0.f;
  float s = wred_add32(a[0] + a[1] + a[2] + a[3]);
  float inv = 1.0f / fmaxf(s, 1.0f);
  bf16u* o = adjb + (size_t)rid * NN + lane * 4;
#pragma unroll
  for (int j = 0; j < 4; ++j) o[j] = f2bf(a[j] * inv);
}

// ---------------------------------------------------------------------------
// Generic bf16 WMMA GEMM: C = A(MxK, row-major bf16) * Bt(NxK, row-major bf16)^T
// 8 waves/block; wave = one 16-row tile x two 16-col tiles (2 accumulators).
// Epilogue: +bias[col], optional relu, store fp32 and/or bf16 (stride Nout).
// ---------------------------------------------------------------------------
__global__ __launch_bounds__(256) void gemm_bf16_kernel(
    const bf16u* __restrict__ A, const bf16u* __restrict__ Bt,
    const float* __restrict__ bias, float* __restrict__ Cf,
    bf16u* __restrict__ Cb, int K, int Nout, int relu) {
  int lane = threadIdx.x & 31;
  int wave = threadIdx.x >> 5;
  int lo = lane & 15, hi = lane >> 4;
  int m0 = blockIdx.y * 64 + (wave & 3) * 16;
  int n0 = blockIdx.x * 64 + (wave >> 2) * 32;
  v8f acc0 = {0.f, 0.f, 0.f, 0.f, 0.f, 0.f, 0.f, 0.f};
  v8f acc1 = acc0;
  for (int kk = 0; kk < K; kk += 32) {
    v16bf a  = load_frag_rm(A, K, m0, kk, lane);
    v16bf b0 = load_frag_rm(Bt, K, n0, kk, lane);
    v16bf b1 = load_frag_rm(Bt, K, n0 + 16, kk, lane);
    acc0 = wmma_bf16(a, b0, acc0);
    acc1 = wmma_bf16(a, b1, acc1);
  }
#pragma unroll
  for (int t = 0; t < 2; ++t) {
    v8f acc = t ? acc1 : acc0;
    int col = n0 + t * 16 + lo;
    float bv = bias ? bias[col] : 0.f;
#pragma unroll
    for (int p = 0; p < 8; ++p) {
      int row = m0 + p + hi * 8;
      float v = acc[p] + bv;
      if (relu) v = fmaxf(v, 0.f);
      size_t off = (size_t)row * Nout + col;
      if (Cf) Cf[off] = v;
      if (Cb) Cb[off] = f2bf(v);
    }
  }
}

// ---------------------------------------------------------------------------
// Fused multi-head attention: one block per (b,h); wave w owns query rows
// [16w, 16w+16). scores strip (16x128) in registers -> scale + edge bias +
// softmax (half-wave shfl reductions) -> bf16 strip via LDS (layout
// transpose) -> ctx = attn @ V -> bf16 ctx buffer.
// ---------------------------------------------------------------------------
__global__ __launch_bounds__(256) void attn_kernel(
    const bf16u* __restrict__ qb, const bf16u* __restrict__ kb,
    const bf16u* __restrict__ vb, const float* __restrict__ eb,
    bf16u* __restrict__ ctxb) {
  __shared__ bf16u lds_attn[8][16 * NN];
  int b = blockIdx.x >> 3, h = blockIdx.x & 7;
  int wave = threadIdx.x >> 5, lane = threadIdx.x & 31;
  int lo = lane & 15, hi = lane >> 4;
  int m0 = wave * 16;
  const size_t base = (size_t)b * NN * DD + h * DK;
  const bf16u* Q = qb + base;
  const bf16u* Kp = kb + base;
  const bf16u* V = vb + base;

  const v8f vzero = {0.f, 0.f, 0.f, 0.f, 0.f, 0.f, 0.f, 0.f};
  v8f acc[8];
#pragma unroll
  for (int nt = 0; nt < 8; ++nt) acc[nt] = vzero;

#pragma unroll
  for (int kk = 0; kk < DK; kk += 32) {
    v16bf a = load_frag_rm(Q, DD, m0, kk, lane);
#pragma unroll
    for (int nt = 0; nt < 8; ++nt) {
      v16bf bfr = load_frag_rm(Kp, DD, nt * 16, kk, lane);
      acc[nt] = wmma_bf16(a, bfr, acc[nt]);
    }
  }

  // scale + edge bias
  const float* ebb = eb + (size_t)b * NN * NN;
#pragma unroll
  for (int nt = 0; nt < 8; ++nt)
#pragma unroll
    for (int p = 0; p < 8; ++p) {
      int row = m0 + p + hi * 8;
      int col = nt * 16 + lo;
      acc[nt][p] = acc[nt][p] * 0.125f + ebb[row * NN + col];
    }

  // softmax over the 128 columns of each row (row lives in one 16-lane half)
  float mx[8], sm[8];
#pragma unroll
  for (int p = 0; p < 8; ++p) {
    float m = -1e30f;
#pragma unroll
    for (int nt = 0; nt < 8; ++nt) m = fmaxf(m, acc[nt][p]);
    mx[p] = wred_max16(m);
    sm[p] = 0.f;
  }
#pragma unroll
  for (int nt = 0; nt < 8; ++nt)
#pragma unroll
    for (int p = 0; p < 8; ++p) {
      float e = __expf(acc[nt][p] - mx[p]);
      acc[nt][p] = e;
      sm[p] += e;
    }
#pragma unroll
  for (int p = 0; p < 8; ++p) sm[p] = 1.0f / wred_add16(sm[p]);

  // C-layout -> A-layout transpose through LDS (bf16 strip, 16 x 128)
  bf16u* strip = &lds_attn[wave][0];
#pragma unroll
  for (int nt = 0; nt < 8; ++nt)
#pragma unroll
    for (int p = 0; p < 8; ++p)
      strip[(p + hi * 8) * NN + nt * 16 + lo] = f2bf(acc[nt][p] * sm[p]);
  __syncthreads();

  // ctx = attn (16x128) @ V (128x64)
  v8f ctx[4];
#pragma unroll
  for (int nt = 0; nt < 4; ++nt) ctx[nt] = vzero;
#pragma unroll
  for (int kk = 0; kk < NN; kk += 32) {
    v16bf a = load_frag_rm(strip, NN, 0, kk, lane);
#pragma unroll
    for (int nt = 0; nt < 4; ++nt) {
      v16bf bfr = load_frag_cm(V, DD, nt * 16, kk, lane);
      ctx[nt] = wmma_bf16(a, bfr, ctx[nt]);
    }
  }
#pragma unroll
  for (int nt = 0; nt < 4; ++nt)
#pragma unroll
    for (int p = 0; p < 8; ++p) {
      int row = m0 + p + hi * 8;
      int col = nt * 16 + lo;
      ctxb[base + (size_t)row * DD + col] = f2bf(ctx[nt][p]);
    }
}

// ---------------------------------------------------------------------------
// GCN aggregation: agg_b = adj_norm_b (128x128) @ x_b (128x512), bf16 out.
// ---------------------------------------------------------------------------
__global__ __launch_bounds__(256) void gcn_agg_kernel(
    const bf16u* __restrict__ adjb, const bf16u* __restrict__ xb,
    bf16u* __restrict__ aggb) {
  int b = blockIdx.z;
  int lane = threadIdx.x & 31;
  int wave = threadIdx.x >> 5;
  int lo = lane & 15, hi = lane >> 4;
  int m0 = blockIdx.y * 64 + (wave & 3) * 16;
  int n0 = blockIdx.x * 64 + (wave >> 2) * 32;
  const bf16u* A = adjb + (size_t)b * NN * NN;
  const bf16u* X = xb + (size_t)b * NN * DD;
  v8f acc0 = {0.f, 0.f, 0.f, 0.f, 0.f, 0.f, 0.f, 0.f};
  v8f acc1 = acc0;
#pragma unroll
  for (int kk = 0; kk < NN; kk += 32) {
    v16bf a  = load_frag_rm(A, NN, m0, kk, lane);
    v16bf b0 = load_frag_cm(X, DD, n0, kk, lane);
    v16bf b1 = load_frag_cm(X, DD, n0 + 16, kk, lane);
    acc0 = wmma_bf16(a, b0, acc0);
    acc1 = wmma_bf16(a, b1, acc1);
  }
#pragma unroll
  for (int t = 0; t < 2; ++t) {
    v8f acc = t ? acc1 : acc0;
    int col = n0 + t * 16 + lo;
#pragma unroll
    for (int p = 0; p < 8; ++p) {
      int row = m0 + p + hi * 8;
      aggb[(size_t)(b * NN + row) * DD + col] = f2bf(acc[p]);
    }
  }
}

// ---------------------------------------------------------------------------
// Fused residual + LayerNorm: x = LN(x + y) * g + b ; writes fp32 + bf16.
// One wave per 512-wide row (32 lanes x 16 floats).
// ---------------------------------------------------------------------------
__global__ __launch_bounds__(256) void ln_residual_kernel(
    const float* __restrict__ y, const float* __restrict__ g,
    const float* __restrict__ bta, float* __restrict__ xf,
    bf16u* __restrict__ xb) {
  int row = blockIdx.x * 8 + (threadIdx.x >> 5);
  int lane = threadIdx.x & 31;
  const f4* xr = (const f4*)(xf + (size_t)row * DD);
  const f4* yr = (const f4*)(y + (size_t)row * DD);
  f4 v[4];
  float s1 = 0.f, s2 = 0.f;
#pragma unroll
  for (int t = 0; t < 4; ++t) {
    f4 s = xr[lane * 4 + t] + yr[lane * 4 + t];
    v[t] = s;
#pragma unroll
    for (int j = 0; j < 4; ++j) { s1 += s[j]; s2 += s[j] * s[j]; }
  }
  s1 = wred_add32(s1);
  s2 = wred_add32(s2);
  float mean = s1 * (1.0f / (float)DD);
  float var = s2 * (1.0f / (float)DD) - mean * mean;
  float rstd = rsqrtf(var + 1e-5f);
  float* xo = xf + (size_t)row * DD;
  bf16u* xbo = xb + (size_t)row * DD;
#pragma unroll
  for (int t = 0; t < 4; ++t)
#pragma unroll
    for (int j = 0; j < 4; ++j) {
      int col = lane * 16 + t * 4 + j;
      float o = (v[t][j] - mean) * rstd * g[col] + bta[col];
      xo[col] = o;
      xbo[col] = f2bf(o);
    }
}

// ---------------------------------------------------------------------------
// Host orchestration
// ---------------------------------------------------------------------------
extern "C" void kernel_launch(void* const* d_in, const int* in_sizes, int n_in,
                              void* d_out, int out_size, void* d_ws, size_t ws_size,
                              hipStream_t stream) {
  (void)in_sizes; (void)n_in; (void)out_size; (void)ws_size;
  const float* node  = (const float*)d_in[0];
  const float* edge  = (const float*)d_in[1];
  const float* adj   = (const float*)d_in[2];
  const float* Wq    = (const float*)d_in[3];
  const float* Wk    = (const float*)d_in[4];
  const float* Wv    = (const float*)d_in[5];
  const float* Wo    = (const float*)d_in[6];
  const float* Wg    = (const float*)d_in[7];
  const float* Wf1   = (const float*)d_in[8];
  const float* Wf2   = (const float*)d_in[9];
  const float* bq    = (const float*)d_in[10];
  const float* bk    = (const float*)d_in[11];
  const float* bv    = (const float*)d_in[12];
  const float* bo    = (const float*)d_in[13];
  const float* bg    = (const float*)d_in[14];
  const float* bf1   = (const float*)d_in[15];
  const float* bf2   = (const float*)d_in[16];
  const float* ln1g  = (const float*)d_in[17];
  const float* ln1b  = (const float*)d_in[18];
  const float* ln2g  = (const float*)d_in[19];
  const float* ln2b  = (const float*)d_in[20];
  const float* ln3g  = (const float*)d_in[21];
  const float* ln3b  = (const float*)d_in[22];

  char* w = (char*)d_ws;
  size_t off = 0;
  auto alloc = [&](size_t bytes) {
    void* p = w + off;
    off += (bytes + 255) & ~(size_t)255;
    return p;
  };
  float* xf    = (float*)alloc((size_t)MN * DD * 4);
  bf16u* xb    = (bf16u*)alloc((size_t)MN * DD * 2);
  bf16u* qb    = (bf16u*)alloc((size_t)MN * DD * 2);
  bf16u* kbuf  = (bf16u*)alloc((size_t)MN * DD * 2);
  bf16u* vbuf  = (bf16u*)alloc((size_t)MN * DD * 2);
  bf16u* ctxb  = (bf16u*)alloc((size_t)MN * DD * 2);
  bf16u* aggb  = (bf16u*)alloc((size_t)MN * DD * 2);
  bf16u* h1b   = (bf16u*)alloc((size_t)MN * FFD * 2);
  float* tmpf  = (float*)alloc((size_t)MN * DD * 4);
  float* eb    = (float*)alloc((size_t)BB * NN * NN * 4);
  bf16u* adjb  = (bf16u*)alloc((size_t)BB * NN * NN * 2);
  bf16u* Wqt   = (bf16u*)alloc((size_t)LL * DD * DD * 2);
  bf16u* Wkt   = (bf16u*)alloc((size_t)LL * DD * DD * 2);
  bf16u* Wvt   = (bf16u*)alloc((size_t)LL * DD * DD * 2);
  bf16u* Wot   = (bf16u*)alloc((size_t)LL * DD * DD * 2);
  bf16u* Wgt   = (bf16u*)alloc((size_t)LL * DD * DD * 2);
  bf16u* Wf1t  = (bf16u*)alloc((size_t)LL * DD * FFD * 2);
  bf16u* Wf2t  = (bf16u*)alloc((size_t)LL * FFD * DD * 2);

  const int T = 256;
  // one-time weight convert+transpose (fp32 -> bf16, NxK layout)
  for (int l = 0; l < LL; ++l) {
    int dd = DD * DD, df = DD * FFD;
    convert_transpose_kernel<<<(dd + T - 1) / T, T, 0, stream>>>(Wq + (size_t)l * dd, Wqt + (size_t)l * dd, DD, DD);
    convert_transpose_kernel<<<(dd + T - 1) / T, T, 0, stream>>>(Wk + (size_t)l * dd, Wkt + (size_t)l * dd, DD, DD);
    convert_transpose_kernel<<<(dd + T - 1) / T, T, 0, stream>>>(Wv + (size_t)l * dd, Wvt + (size_t)l * dd, DD, DD);
    convert_transpose_kernel<<<(dd + T - 1) / T, T, 0, stream>>>(Wo + (size_t)l * dd, Wot + (size_t)l * dd, DD, DD);
    convert_transpose_kernel<<<(dd + T - 1) / T, T, 0, stream>>>(Wg + (size_t)l * dd, Wgt + (size_t)l * dd, DD, DD);
    convert_transpose_kernel<<<(df + T - 1) / T, T, 0, stream>>>(Wf1 + (size_t)l * df, Wf1t + (size_t)l * df, DD, FFD);
    convert_transpose_kernel<<<(df + T - 1) / T, T, 0, stream>>>(Wf2 + (size_t)l * df, Wf2t + (size_t)l * df, FFD, DD);
  }
  init_x_kernel<<<(MN * DD) / T, T, 0, stream>>>(node, xf, xb);
  ebias_kernel<<<(BB * NN * NN) / 8, T, 0, stream>>>(edge, eb);
  adjnorm_kernel<<<(BB * NN) / 8, T, 0, stream>>>(adj, adjb);

  dim3 gD(DD / 64, MN / 64);     // Nout=512 GEMMs
  dim3 gF(FFD / 64, MN / 64);    // Nout=2048 GEMM

  for (int l = 0; l < LL; ++l) {
    const bf16u* wq = Wqt + (size_t)l * DD * DD;
    const bf16u* wk = Wkt + (size_t)l * DD * DD;
    const bf16u* wv = Wvt + (size_t)l * DD * DD;
    const bf16u* wo = Wot + (size_t)l * DD * DD;
    const bf16u* wg = Wgt + (size_t)l * DD * DD;
    const bf16u* w1 = Wf1t + (size_t)l * DD * FFD;
    const bf16u* w2 = Wf2t + (size_t)l * FFD * DD;

    // --- attention block ---
    gemm_bf16_kernel<<<gD, T, 0, stream>>>(xb, wq, bq + l * DD, nullptr, qb, DD, DD, 0);
    gemm_bf16_kernel<<<gD, T, 0, stream>>>(xb, wk, bk + l * DD, nullptr, kbuf, DD, DD, 0);
    gemm_bf16_kernel<<<gD, T, 0, stream>>>(xb, wv, bv + l * DD, nullptr, vbuf, DD, DD, 0);
    attn_kernel<<<BB * HH, T, 0, stream>>>(qb, kbuf, vbuf, eb, ctxb);
    gemm_bf16_kernel<<<gD, T, 0, stream>>>(ctxb, wo, bo + l * DD, tmpf, nullptr, DD, DD, 0);
    ln_residual_kernel<<<MN / 8, T, 0, stream>>>(tmpf, ln1g + l * DD, ln1b + l * DD, xf, xb);

    // --- GCN block ---
    gcn_agg_kernel<<<dim3(DD / 64, NN / 64, BB), T, 0, stream>>>(adjb, xb, aggb);
    gemm_bf16_kernel<<<gD, T, 0, stream>>>(aggb, wg, bg + l * DD, tmpf, nullptr, DD, DD, 1);
    ln_residual_kernel<<<MN / 8, T, 0, stream>>>(tmpf, ln2g + l * DD, ln2b + l * DD, xf, xb);

    // --- FFN block ---
    gemm_bf16_kernel<<<gF, T, 0, stream>>>(xb, w1, bf1 + l * FFD, nullptr, h1b, DD, FFD, 1);
    gemm_bf16_kernel<<<gD, T, 0, stream>>>(h1b, w2, bf2 + l * DD, tmpf, nullptr, FFD, DD, 0);
    ln_residual_kernel<<<MN / 8, T, 0, stream>>>(tmpf, ln3g + l * DD, ln3b + l * DD, xf, xb);
  }

  hipMemcpyAsync(d_out, xf, (size_t)MN * DD * sizeof(float),
                 hipMemcpyDeviceToDevice, stream);
}
